// IssLSTM_76321568850362
// MI455X (gfx1250) — compile-verified, matrix operand order checked by reading, not run on hardware
//
#include <hip/hip_runtime.h>
#include <hip/hip_bf16.h>

typedef __bf16 bf16_t;
typedef bf16_t v16bf __attribute__((ext_vector_type(16)));
typedef bf16_t v8bf  __attribute__((ext_vector_type(8)));
typedef float  v8f   __attribute__((ext_vector_type(8)));

constexpr int kB = 32;     // batch
constexpr int kT = 2048;   // time steps
constexpr int kI = 64;     // input dim
constexpr int kH = 512;    // hidden
constexpr int kO = 64;     // output dim
constexpr int kK = kH + kI;        // 576 fused K dim [h | u_t]
constexpr int kKS = kK / 32;       // 18 k-steps
constexpr int kNB = 16;            // blocks in scan kernel
constexpr int kJPB = kH / kNB;     // 32 hidden units per block
constexpr int kWST = 584;          // LDS row stride (halves), 16B aligned, conflict-free
constexpr int kAST = 584;

struct SmemScan {
    bf16_t W[4 * kJPB][kWST];   // 128 x 584 bf16 = 149504 B (weights, resident)
    bf16_t A[kB][kAST];         //  32 x 584 bf16 =  37376 B ([h_t | u_t] staging)
    float  bias[4 * kJPB];      //  512 B (b_ih + b_hh for owned gate rows)
    float  cell[64][20];        //  5120 B: per-thread c-state (16 used, pad 20)
};

struct SmemOut {
    bf16_t W[kO][520];          // 64 x 520 bf16 = 66560 B (w_lin)
};

__device__ __forceinline__ float sigm(float x) {
    return 1.0f / (1.0f + __expf(-x));
}

// Load a 16x32 bf16 WMMA A/B fragment from a row-major [16][stride] tile.
// Lane L<16: row L, K = {0..7, 16..23}; lane L+16: row L-16, K = {8..15, 24..31}.
__device__ __forceinline__ v16bf ld_frag(const bf16_t* tile, int stride, int lane, int k0) {
    const bf16_t* p = tile + (size_t)(lane & 15) * stride + k0 + ((lane >> 4) << 3);
    v8bf lo = *(const v8bf*)(p);
    v8bf hi = *(const v8bf*)(p + 16);
    return __builtin_shufflevector(lo, hi, 0,1,2,3,4,5,6,7,8,9,10,11,12,13,14,15);
}

// Device-wide generation barrier (16 blocks). Counters zeroed by hipMemsetAsync per launch.
__device__ __forceinline__ void grid_barrier(unsigned* cnt, unsigned* gen,
                                             unsigned target, unsigned nblk) {
    __threadfence();          // each wave's stores reach device scope
    __syncthreads();
    if (threadIdx.x == 0) {
        unsigned old = __hip_atomic_fetch_add(cnt, 1u, __ATOMIC_ACQ_REL,
                                              __HIP_MEMORY_SCOPE_AGENT);
        if (old == nblk - 1u) {
            __hip_atomic_store(cnt, 0u, __ATOMIC_RELAXED, __HIP_MEMORY_SCOPE_AGENT);
            __hip_atomic_store(gen, target, __ATOMIC_RELEASE, __HIP_MEMORY_SCOPE_AGENT);
        } else {
            while (__hip_atomic_load(gen, __ATOMIC_ACQUIRE,
                                     __HIP_MEMORY_SCOPE_AGENT) < target) {
                __builtin_amdgcn_s_sleep(2);
            }
        }
    }
    __syncthreads();
    __builtin_amdgcn_fence(__ATOMIC_ACQUIRE, "agent");  // invalidate stale WGP$ lines
}

__global__ __launch_bounds__(64)
__attribute__((amdgpu_waves_per_eu(1)))   // 2 waves/block, 16 blocks: max VGPR budget
void lstm_scan_kernel(
    const float* __restrict__ u, const float* __restrict__ h0,
    const float* __restrict__ c0, const float* __restrict__ w_ih,
    const float* __restrict__ w_hh, const float* __restrict__ b_ih,
    const float* __restrict__ b_hh, bf16_t* __restrict__ hs,
    float* __restrict__ hT, float* __restrict__ cT,
    unsigned* cnt, unsigned* gen) {

    extern __shared__ char smem_raw[];
    SmemScan& S = *(SmemScan*)smem_raw;

    const int tid  = threadIdx.x;
    const int lane = tid & 31;
    const int wv   = tid >> 5;        // wave 0: batches 0-15, wave 1: 16-31
    const int m0   = wv * 16;
    const int bsel = lane >> 4;       // +8 rows for upper half-wave in D layout
    const int j0   = blockIdx.x * kJPB;

    // ---- One-time: convert owned weight rows f32 -> bf16 into LDS ----
    // Row rr = gate*32 + jl  (gate order i,f,g,o), K = [w_hh(512) | w_ih(64)].
    for (int rr = 0; rr < 4 * kJPB; ++rr) {
        const int gate = rr >> 5;
        const int jl   = rr & 31;
        const int g    = gate * kH + j0 + jl;
        for (int k = tid; k < kK; k += 64) {
            float v = (k < kH) ? w_hh[(size_t)g * kH + k]
                               : w_ih[(size_t)g * kI + (k - kH)];
            S.W[rr][k] = (bf16_t)v;
        }
    }
    for (int rr = tid; rr < 4 * kJPB; rr += 64) {
        const int gate = rr >> 5;
        const int g    = gate * kH + j0 + (rr & 31);
        S.bias[rr] = b_ih[g] + b_hh[g];
    }

    // ---- Init hs[0] slice (bf16 broadcast buffer) ----
    for (int idx = tid; idx < kB * kJPB; idx += 64) {
        const int b = idx >> 5;
        const int j = j0 + (idx & 31);
        hs[(size_t)b * kH + j] = (bf16_t)h0[b * kH + j];
    }

    // ---- Init per-thread cell state in LDS:
    //      cell[tid][js*8 + r] <-> (b = m0+r+8*bsel, j = j0+js*16+(lane&15)) ----
    #pragma unroll
    for (int r = 0; r < 8; ++r) {
        const int b = m0 + r + bsel * 8;
        #pragma unroll
        for (int js = 0; js < 2; ++js) {
            const int j = j0 + js * 16 + (lane & 15);
            S.cell[tid][js * 8 + r] = c0[b * kH + j];
        }
    }

    grid_barrier(cnt, gen, 1u, gridDim.x);

    for (int t = 0; t < kT; ++t) {
        // ---- Stage A = [h_t (bf16, from global ring) | u_t (f32 -> bf16)] ----
        for (int idx = tid; idx < kB * 64; idx += 64) {       // 2048 x 16B chunks
            const int b  = idx >> 6;
            const int co = idx & 63;
            const uint4* src = (const uint4*)(hs + (size_t)(t * kB + b) * kH);
            *(uint4*)(&S.A[b][co * 8]) = src[co];
        }
        for (int idx = tid; idx < kB * kI; idx += 64) {
            const int b = idx >> 6;
            const int i = idx & 63;
            S.A[b][kH + i] = (bf16_t)u[((size_t)b * kT + t) * kI + i];
        }
        // Prefetch next timestep's u rows into cache (global_prefetch_b8);
        // 2 cachelines per row of 256 B.
        if (t + 1 < kT) {
            const int pb = tid >> 1;          // row (batch) 0..31
            const int ph = tid & 1;           // half-row
            __builtin_prefetch(&u[((size_t)pb * kT + (t + 1)) * kI + ph * 32], 0, 0);
        }
        __syncthreads();

        // ---- gates[32, 8x16] = A[32,576] x W^T, bf16 WMMA, f32 accumulate ----
        // Structurally pipelined: two live groups of 4 B-fragments (forced into
        // distinct registers), loads for the next group issued a full 4-WMMA
        // phase ahead; sched_barrier(0) keeps the scheduler from re-gluing
        // loads onto their consumers -> partial s_wait_dscnt instead of 0.
        v8f acc[8];
        #pragma unroll
        for (int n = 0; n < 8; ++n) {
            const float bv = S.bias[n * 16 + (lane & 15)];   // short-lived temp
            #pragma unroll
            for (int r = 0; r < 8; ++r) acc[n][r] = bv;
        }

        v16bf afd[2];
        v16bf bg0[4], bg1[4];
        afd[0] = ld_frag(&S.A[m0][0], kAST, lane, 0);
        #pragma unroll
        for (int n = 0; n < 4; ++n)
            bg0[n] = ld_frag(&S.W[n * 16][0], kWST, lane, 0);

        for (int kk = 0; kk < kKS; ++kk) {
            const int k0 = kk * 32;
            const v16bf af = afd[kk & 1];

            // Phase 0: load tiles 4..7 @ k0 (+ next A), compute tiles 0..3 @ k0
            #pragma unroll
            for (int n = 0; n < 4; ++n)
                bg1[n] = ld_frag(&S.W[(4 + n) * 16][0], kWST, lane, k0);
            if (kk + 1 < kKS)
                afd[(kk + 1) & 1] = ld_frag(&S.A[m0][0], kAST, lane, k0 + 32);
            __builtin_amdgcn_sched_barrier(0);
            #pragma unroll
            for (int n = 0; n < 4; ++n)
                acc[n] = __builtin_amdgcn_wmma_f32_16x16x32_bf16(
                    false, af, false, bg0[n], (short)0, acc[n], false, false);
            __builtin_amdgcn_sched_barrier(0);

            // Phase 1: load tiles 0..3 @ k0+32, compute tiles 4..7 @ k0
            if (kk + 1 < kKS) {
                #pragma unroll
                for (int n = 0; n < 4; ++n)
                    bg0[n] = ld_frag(&S.W[n * 16][0], kWST, lane, k0 + 32);
            }
            __builtin_amdgcn_sched_barrier(0);
            #pragma unroll
            for (int n = 0; n < 4; ++n)
                acc[4 + n] = __builtin_amdgcn_wmma_f32_16x16x32_bf16(
                    false, af, false, bg1[n], (short)0, acc[4 + n], false, false);
            __builtin_amdgcn_sched_barrier(0);
        }

        // ---- Activations + state update (c-state lives in LDS, not VGPRs);
        //      write h_{t+1} slice (bf16) ----
        #pragma unroll
        for (int r = 0; r < 8; ++r) {
            const int b = m0 + r + bsel * 8;
            #pragma unroll
            for (int js = 0; js < 2; ++js) {
                const float iv = sigm(acc[0 + js][r]);
                const float fv = sigm(acc[2 + js][r]);
                const float gv = tanhf(acc[4 + js][r]);
                const float ov = sigm(acc[6 + js][r]);
                const float cold = S.cell[tid][js * 8 + r];
                const float cv = fv * cold + iv * gv;
                S.cell[tid][js * 8 + r] = cv;
                const float hv = ov * tanhf(cv);
                const int j = j0 + js * 16 + (lane & 15);
                hs[(size_t)((t + 1) * kB + b) * kH + j] = (bf16_t)hv;
                if (t == kT - 1) {
                    hT[b * kH + j] = hv;
                    cT[b * kH + j] = cv;
                }
            }
        }

        grid_barrier(cnt, gen, (unsigned)(t + 2), gridDim.x);
    }
}

__global__ __launch_bounds__(256) void out_gemm_kernel(
    const bf16_t* __restrict__ hs, const float* __restrict__ w_lin,
    const float* __restrict__ b_lin, float* __restrict__ out) {

    extern __shared__ char smem_raw[];
    SmemOut& S = *(SmemOut*)smem_raw;

    for (int idx = threadIdx.x; idx < kO * kH; idx += 256) {
        const int o = idx >> 9;
        const int k = idx & 511;
        S.W[o][k] = (bf16_t)w_lin[(size_t)o * kH + k];
    }
    __syncthreads();

    const int lane = threadIdx.x & 31;
    const int wv   = threadIdx.x >> 5;
    const int mt   = blockIdx.x * 8 + wv;        // M-tile over flattened (t,b) rows
    const bf16_t* arow = hs + (size_t)(mt * 16 + kB) * kH;   // skip hs[0] (= h0)

    float bl[4];
    #pragma unroll
    for (int n = 0; n < 4; ++n) bl[n] = b_lin[n * 16 + (lane & 15)];
    v8f acc[4];
    #pragma unroll
    for (int n = 0; n < 4; ++n) {
        #pragma unroll
        for (int r = 0; r < 8; ++r) acc[n][r] = bl[n];
    }

    // Double-buffered B set + global A fragment one full k-step ahead.
    v16bf afd[2];
    v16bf bb[2][4];
    afd[0] = ld_frag(arow, kH, lane, 0);
    #pragma unroll
    for (int n = 0; n < 4; ++n)
        bb[0][n] = ld_frag(&S.W[n * 16][0], 520, lane, 0);

    for (int kk = 0; kk < kH / 32; ++kk) {        // 16 k-steps
        const int k0 = kk * 32;
        const int cur = kk & 1;
        if (kk + 1 < kH / 32) {
            afd[cur ^ 1] = ld_frag(arow, kH, lane, k0 + 32);
            #pragma unroll
            for (int n = 0; n < 4; ++n)
                bb[cur ^ 1][n] = ld_frag(&S.W[n * 16][0], 520, lane, k0 + 32);
        }
        __builtin_amdgcn_sched_barrier(0);
        #pragma unroll
        for (int n = 0; n < 4; ++n)
            acc[n] = __builtin_amdgcn_wmma_f32_16x16x32_bf16(
                false, afd[cur], false, bb[cur][n], (short)0, acc[n], false, false);
        __builtin_amdgcn_sched_barrier(0);
    }

    const int t    = mt >> 1;
    const int b0   = (mt & 1) * 16;
    const int bsel = lane >> 4;
    #pragma unroll
    for (int n = 0; n < 4; ++n) {
        const int o = n * 16 + (lane & 15);
        #pragma unroll
        for (int r = 0; r < 8; ++r) {
            const int b = b0 + r + bsel * 8;
            out[((size_t)b * kT + t) * kO + o] = acc[n][r];
        }
    }
}

extern "C" void kernel_launch(void* const* d_in, const int* in_sizes, int n_in,
                              void* d_out, int out_size, void* d_ws, size_t ws_size,
                              hipStream_t stream) {
    (void)in_sizes; (void)n_in; (void)out_size; (void)ws_size;

    const float* u     = (const float*)d_in[0];
    const float* h0    = (const float*)d_in[1];
    const float* c0    = (const float*)d_in[2];
    const float* w_ih  = (const float*)d_in[3];
    const float* w_hh  = (const float*)d_in[4];
    const float* b_ih  = (const float*)d_in[5];
    const float* b_hh  = (const float*)d_in[6];
    const float* w_lin = (const float*)d_in[7];
    const float* b_lin = (const float*)d_in[8];

    float* out = (float*)d_out;                       // [B,T,O]
    float* hT  = out + (size_t)kB * kT * kO;          // [1,B,H]
    float* cT  = hT + (size_t)kB * kH;                // [1,B,H]

    // Workspace: [0,256) sync counters, then hs ring [(T+1)*B][H] bf16 (~64 MB)
    unsigned* cnt = (unsigned*)d_ws;
    unsigned* gen = cnt + 1;
    bf16_t* hs = (bf16_t*)((char*)d_ws + 256);

    hipMemsetAsync(d_ws, 0, 256, stream);             // reset barrier state (capture-safe)

    lstm_scan_kernel<<<kNB, 64, sizeof(SmemScan), stream>>>(
        u, h0, c0, w_ih, w_hh, b_ih, b_hh, hs, hT, cT, cnt, gen);

    out_gemm_kernel<<<(kT * kB / 16) / 8, 256, sizeof(SmemOut), stream>>>(
        hs, w_lin, b_lin, out);
}
